// DFCCrossCoder_76381698392291
// MI455X (gfx1250) — matrix operand order repaired and checked by reading, not compiled, not run on hardware
//
#include <hip/hip_runtime.h>
#include <cstddef>

// ---------------------------------------------------------------------------
// Types for CDNA5 WMMA (wave32): v16bf A/B fragments, v8f f32 accumulator.
// ---------------------------------------------------------------------------
typedef __attribute__((ext_vector_type(16))) __bf16       v16bf;
typedef __attribute__((ext_vector_type(8)))  float        v8f;
typedef __attribute__((ext_vector_type(4)))  float        v4f;
typedef __attribute__((ext_vector_type(4)))  unsigned int v4u;

union FragAB { v4u q[2]; v16bf v; };

__device__ __forceinline__ unsigned short f2bf(float f) {
  // round-to-nearest-even fp32 -> bf16
  unsigned u = __float_as_uint(f);
  u += 0x7FFFu + ((u >> 16) & 1u);
  return (unsigned short)(u >> 16);
}
__device__ __forceinline__ unsigned packbf2(float lo, float hi) {
  return (unsigned)f2bf(lo) | ((unsigned)f2bf(hi) << 16);
}

// ---------------------------------------------------------------------------
// Tiled bf16 WMMA GEMM:  C[M,N] = act( A[M,K] * (B[K,N] .* mask) + bias[N] )
//   Block tile 64(M) x 256(N), K staged 64 per barrier, 256 threads = 8 waves
//   arranged 2(M) x 4(N); each wave owns a 32x64 C tile -> 16 x
//   v_wmma_f32_16x16x32_bf16 per stage (2 mi x 4 ni x 2 k-substeps).
// B tile is stored transposed in LDS as Bt[n][k] (bf16), with the 8-half
//   K-granule XOR-swizzled by (n>>5)&7 so the b32 pack-stores are bank-
//   conflict-free while fragment reads remain contiguous ds_load_b128.
// Requires: M%64==0, N%256==0, K%64==0 (true for all calls below).
// ---------------------------------------------------------------------------
#define BM 64
#define BN 256
#define BK 64
#define AS_STRIDE 72   // halves per A-tile LDS row (144B, 16B aligned)
#define BS_STRIDE 72   // halves per Bt-tile LDS row

__global__ __launch_bounds__(256)
void gemm_bf16_wmma(const float* __restrict__ A, const float* __restrict__ B,
                    const float* __restrict__ bias, const float* __restrict__ mask,
                    float* __restrict__ C, int M, int N, int K,
                    int maskSplit, int doRelu) {
  __shared__ unsigned short As[BM * AS_STRIDE];   //  9216 B
  __shared__ unsigned short Bt[BN * BS_STRIDE];   // 36864 B  ([n][k], swizzled)

  const int tid   = threadIdx.x;
  const int lane  = tid & 31;
  const int wid   = tid >> 5;
  const int wm    = wid >> 2;        // 0..1 : 32-row slab within 64
  const int wn    = wid & 3;         // 0..3 : 64-col slab within 256
  const int lr    = lane & 15;       // row/col-within-16 for fragments
  const int khalf = lane >> 4;       // 0/1 : which K half this lane holds

  const int bm = blockIdx.y * BM;
  const int bn = blockIdx.x * BN;

  // A loader: thread -> row ar (0..63), 16 consecutive k at ac
  const int ar = tid >> 2;
  const int ac = (tid & 3) * 16;
  // B loader: thread -> k-row pair kp (0..31), 32 consecutive n at nb0
  const int kp  = tid >> 3;
  const int nb0 = (tid & 7) * 32;

  v8f acc[2][4];
#pragma unroll
  for (int mi = 0; mi < 2; ++mi)
#pragma unroll
    for (int ni = 0; ni < 4; ++ni)
      acc[mi][ni] = (v8f){0.f, 0.f, 0.f, 0.f, 0.f, 0.f, 0.f, 0.f};

  for (int k0 = 0; k0 < K; k0 += BK) {
    // ---- stage A tile (64x64 f32 -> bf16, row-major) ----
    {
      const float* ap = A + (size_t)(bm + ar) * K + (k0 + ac);
      v4f a0 = *(const v4f*)ap;
      v4f a1 = *(const v4f*)(ap + 4);
      v4f a2 = *(const v4f*)(ap + 8);
      v4f a3 = *(const v4f*)(ap + 12);
      v4u w0, w1;
      w0.x = packbf2(a0[0], a0[1]);  w0.y = packbf2(a0[2], a0[3]);
      w0.z = packbf2(a1[0], a1[1]);  w0.w = packbf2(a1[2], a1[3]);
      w1.x = packbf2(a2[0], a2[1]);  w1.y = packbf2(a2[2], a2[3]);
      w1.z = packbf2(a3[0], a3[1]);  w1.w = packbf2(a3[2], a3[3]);
      *(v4u*)&As[ar * AS_STRIDE + ac]     = w0;
      *(v4u*)&As[ar * AS_STRIDE + ac + 8] = w1;
    }
    // ---- stage B tile (64x256 f32 -> bf16, transposed + swizzled) ----
    {
      const int krow = k0 + 2 * kp;
      const float* bp0 = B + (size_t)krow * N + bn + nb0;
      const float* bp1 = bp0 + N;
      float m00 = 1.f, m01 = 1.f, m10 = 1.f, m11 = 1.f;
      if (mask) {
        m00 = mask[krow * 2];       m01 = mask[krow * 2 + 1];
        m10 = mask[krow * 2 + 2];   m11 = mask[krow * 2 + 3];
      }
      const int g    = kp >> 2;          // K granule (8 halves) within row
      const int intr = 2 * (kp & 3);     // half offset within granule
#pragma unroll
      for (int c = 0; c < 8; ++c) {
        v4f r0 = *(const v4f*)(bp0 + c * 4);
        v4f r1 = *(const v4f*)(bp1 + c * 4);
#pragma unroll
        for (int e = 0; e < 4; ++e) {
          const int nn = nb0 + c * 4 + e;                 // block-local column
          const bool hi = (bn + nn) >= maskSplit;
          const float v0 = r0[e] * (hi ? m01 : m00);
          const float v1 = r1[e] * (hi ? m11 : m10);
          const int h = (((g ^ ((nn >> 5) & 7)) << 3) | intr);  // swizzled half
          *(unsigned*)&Bt[nn * BS_STRIDE + h] = packbf2(v0, v1);
        }
      }
    }
    __syncthreads();

    // ---- 2 K-substeps of 32, per ISA 16-bit fragment layouts ----
#pragma unroll
    for (int ks = 0; ks < 2; ++ks) {
      FragAB af[2], bfr[4];
#pragma unroll
      for (int mi = 0; mi < 2; ++mi) {
        const unsigned short* p = &As[(wm * 32 + mi * 16 + lr) * AS_STRIDE + ks * 32];
        af[mi].q[0] = *(const v4u*)(p + khalf * 8);        // K 0..7  / 8..15
        af[mi].q[1] = *(const v4u*)(p + 16 + khalf * 8);   // K 16..23 / 24..31
      }
#pragma unroll
      for (int ni = 0; ni < 4; ++ni) {
        const int ncol = wn * 64 + ni * 16;                // 16-aligned, lr<16: no carry
        const int nsw  = (ncol >> 5) & 7;                  // swizzle key for this tile
        const unsigned short* rowb = &Bt[(ncol + lr) * BS_STRIDE];
        const int gr = ks * 4 + khalf;                     // unswizzled granule
        bfr[ni].q[0] = *(const v4u*)(rowb + (((gr    ) ^ nsw) << 3));
        bfr[ni].q[1] = *(const v4u*)(rowb + (((gr + 2) ^ nsw) << 3));
      }
#pragma unroll
      for (int mi = 0; mi < 2; ++mi)
#pragma unroll
        for (int ni = 0; ni < 4; ++ni)
          acc[mi][ni] = __builtin_amdgcn_wmma_f32_16x16x32_bf16(
              false, af[mi].v, false, bfr[ni].v,
              (short)0, acc[mi][ni], false, false);
    }
    __syncthreads();
  }

  // ---- epilogue: C/D layout: n = lane%16, M = v + 8*(lane>>4) ----
#pragma unroll
  for (int mi = 0; mi < 2; ++mi) {
#pragma unroll
    for (int ni = 0; ni < 4; ++ni) {
      const int n  = bn + wn * 64 + ni * 16 + lr;
      const float bv = bias ? bias[n] : 0.f;
      const int mbase = bm + wm * 32 + mi * 16 + khalf * 8;
#pragma unroll
      for (int v = 0; v < 8; ++v) {
        float r = acc[mi][ni][v] + bv;
        if (doRelu) r = fmaxf(r, 0.f);
        C[(size_t)(mbase + v) * N + n] = r;
      }
    }
  }
}

// ---------------------------------------------------------------------------
// Exact top-K (K=64) per row via 4-pass radix select on float bit patterns
// (valid: post-ReLU values are all >= 0, so uint order == float order).
// Zeroes everything below the 64th-largest; atomically keeps the exact
// number of threshold-equal elements. One 256-thread block per row.
// ---------------------------------------------------------------------------
#define F_DIM 16384
#define TOPK  64

__global__ __launch_bounds__(256)
void topk_select(float* __restrict__ feat) {
  __shared__ unsigned hist[256];
  __shared__ unsigned sh_prefix, sh_need, sh_eq;

  const int tid = threadIdx.x;
  float* rowp = feat + (size_t)blockIdx.x * F_DIM;

  if (tid == 0) { sh_prefix = 0u; sh_need = TOPK; sh_eq = 0u; }
  __syncthreads();

  for (int pass = 0; pass < 4; ++pass) {
    const int shift = 24 - 8 * pass;
    hist[tid] = 0u;
    __syncthreads();
    const unsigned prefix = sh_prefix;
    const unsigned himask = (pass == 0) ? 0u : (0xFFFFFFFFu << (8 * (4 - pass)));
    for (int i = tid; i < F_DIM; i += 256) {
      const unsigned u = __float_as_uint(rowp[i]);
      if ((u & himask) == prefix)
        atomicAdd(&hist[(u >> shift) & 255u], 1u);
    }
    __syncthreads();
    if (tid == 0) {
      const unsigned need = sh_need;
      unsigned cum = 0;
      int b = 255;
      for (; b >= 0; --b) {
        const unsigned c = hist[b];
        if (cum + c >= need) break;
        cum += c;
      }
      sh_need = need - cum;                        // rank within the tie bin
      sh_prefix = prefix | ((unsigned)b << shift); // extend threshold pattern
    }
    __syncthreads();
  }

  const unsigned tau = sh_prefix;
  const unsigned keepEq = sh_need;
  for (int i = tid; i < F_DIM; i += 256) {
    const unsigned u = __float_as_uint(rowp[i]);
    bool keep;
    if (u > tau)       keep = true;
    else if (u == tau) keep = (atomicAdd(&sh_eq, 1u) < keepEq);
    else               keep = false;
    if (!keep) rowp[i] = 0.0f;
  }
}

// ---------------------------------------------------------------------------
// Launch: encode GEMM -> top-k -> decode GEMM.
// d_in: x(4096,2,2048) W_enc(2,2048,16384) b_enc(16384) W_dec(16384,2,2048)
//       b_dec(2,2048) dec_mask(16384,2)  — all float32.
// d_out: [recon 4096*4096][features 4096*16384] float32.
// ---------------------------------------------------------------------------
extern "C" void kernel_launch(void* const* d_in, const int* in_sizes, int n_in,
                              void* d_out, int out_size, void* d_ws, size_t ws_size,
                              hipStream_t stream) {
  (void)in_sizes; (void)n_in; (void)out_size; (void)d_ws; (void)ws_size;

  const float* x     = (const float*)d_in[0];
  const float* W_enc = (const float*)d_in[1];
  const float* b_enc = (const float*)d_in[2];
  const float* W_dec = (const float*)d_in[3];
  const float* b_dec = (const float*)d_in[4];
  const float* dmask = (const float*)d_in[5];

  const int Bv = 4096;   // batch
  const int D2 = 4096;   // 2 * ACT_DIM (flattened m,d)
  const int F  = F_DIM;  // dictionary size

  float* out   = (float*)d_out;
  float* recon = out;                          // (B, 2, 2048) flat = B*D2
  float* feat  = out + (size_t)Bv * D2;        // (B, F)

  dim3 blk(256);

  // encode: pre = relu(x @ W_enc + b_enc), staged into features region
  gemm_bf16_wmma<<<dim3(F / BN, Bv / BM), blk, 0, stream>>>(
      x, W_enc, b_enc, /*mask=*/nullptr, feat, Bv, F, D2, /*maskSplit=*/0, /*relu=*/1);

  // exact top-64 per row, in place
  topk_select<<<dim3(Bv), blk, 0, stream>>>(feat);

  // decode: recon = features @ (W_dec .* dec_mask) + b_dec
  gemm_bf16_wmma<<<dim3(D2 / BN, Bv / BM), blk, 0, stream>>>(
      feat, W_dec, b_dec, /*mask=*/dmask, recon, Bv, D2, F, /*maskSplit=*/2048, /*relu=*/0);
}
